// SparseAxialCausalAttention_51986284151358
// MI455X (gfx1250) — matrix operand order repaired and checked by reading, not compiled, not run on hardware
//
#include <hip/hip_runtime.h>
#include <hip/hip_bf16.h>

// ---------------------------------------------------------------------------
// Sparse axial causal attention (DALL-E style) for MI455X / gfx1250.
// All matmuls run through v_wmma_f32_16x16x32_bf16 (bf16 A/B, f32 accum).
//
// Shapes (from reference): B=8, SEQ=4223 (padded to 4224), DIM=1024,
// H=16, Dh=64, IMG=64, img_seq=4096, text_len=128.
// ---------------------------------------------------------------------------

typedef __bf16 bf16;
typedef __attribute__((ext_vector_type(16))) __bf16 v16bf;
typedef __attribute__((ext_vector_type(8)))  __bf16 v8bf;
typedef __attribute__((ext_vector_type(8)))  float  v8f;

#define B_       8
#define SEQ_     4223
#define SP_      4224          // padded seq (multiple of 64)
#define DIM_     1024
#define H_       16
#define DH_      64
#define TEXT_    128
#define IMG_     64
#define ROWS_    (B_ * SP_)    // 33792
#define NEGF_    (-3.4028234663852886e38f)

// D = A(16x32 bf16) * B(32x16 bf16) + C(16x16 f32)
__device__ __forceinline__ v8f wmma_bf16(v16bf a, v16bf b, v8f c) {
    return __builtin_amdgcn_wmma_f32_16x16x32_bf16(
        /*neg_a=*/false, a, /*neg_b=*/false, b,
        /*c_mod=*/(short)0, c, /*reuse_a=*/false, /*reuse_b=*/false);
}

// A-fragment (16x32 bf16): lane holds row m = lane%16.
// Element e: K = e + 8*g (e<8) ; K = e+8 + 8*g (e>=8), g = lane/16.
// => two contiguous 8-element (16B) loads from the row.
__device__ __forceinline__ v16bf load_a_frag(const bf16* rowptr, int k0, int g) {
    union { v16bf v; v8bf h[2]; } u;
    u.h[0] = *(const v8bf*)(rowptr + k0 + g * 8);
    u.h[1] = *(const v8bf*)(rowptr + k0 + g * 8 + 16);
    return u.v;
}

// B-fragment (32x16 bf16): lane holds col n = lane%16, same K order as A.
// Source is K-major (i.e. a row of the transposed matrix) => two 16B loads.
__device__ __forceinline__ v16bf load_b_frag(const bf16* btrow, int k0, int g) {
    union { v16bf v; v8bf h[2]; } u;
    u.h[0] = *(const v8bf*)(btrow + k0 + g * 8);
    u.h[1] = *(const v8bf*)(btrow + k0 + g * 8 + 16);
    return u.v;
}

// ---------------------------------------------------------------------------
// Stage 1: conversions
// ---------------------------------------------------------------------------
__global__ void cvt_x_kernel(const float* __restrict__ x, bf16* __restrict__ X) {
    size_t i = (size_t)blockIdx.x * blockDim.x + threadIdx.x;   // over ROWS_*DIM_
    if (i >= (size_t)ROWS_ * DIM_) return;
    size_t row = i / DIM_;  int col = (int)(i % DIM_);
    int b = (int)(row / SP_), s = (int)(row % SP_);
    X[i] = (s < SEQ_) ? (bf16)x[((size_t)b * SEQ_ + s) * DIM_ + col] : (bf16)0.0f;
}

__global__ void cvt_wT_kernel(const float* __restrict__ w, bf16* __restrict__ WT,
                              int K, int N) {     // w:[K][N] -> WT:[N][K]
    size_t i = (size_t)blockIdx.x * blockDim.x + threadIdx.x;
    if (i >= (size_t)K * N) return;
    int k = (int)(i / N), n = (int)(i % N);
    WT[(size_t)n * K + k] = (bf16)w[i];
}

// ---------------------------------------------------------------------------
// Stage 2: QKV projection.  C[33792,3072] = X[33792,1024] @ Wqkv.
// Wave tile = 64 rows x 64 cols: per K-step 4 A-frags + 4 B-frags feed
// 16 WMMAs (512 B loaded per WMMA).  Accum = 4x4 v8f = 128 VGPRs;
// __launch_bounds__(256,1) gives the allocator headroom (no spills).
// Row offsets are compile-time (i*16*DIM_) so loads use immediate offsets
// off two base pointers.  Epilogue: 64-row tile never crosses a batch
// boundary (SP_ % 64 == 0), so b / base-s are hoisted per wave.
// ---------------------------------------------------------------------------
__global__ void __launch_bounds__(256, 1)
qkv_gemm_kernel(const bf16* __restrict__ X,
                const bf16* __restrict__ WT,   // [3072][1024]
                bf16* __restrict__ Q, bf16* __restrict__ K,
                bf16* __restrict__ V) {
    const int wave = threadIdx.x >> 5, lane = threadIdx.x & 31;
    const int g = lane >> 4, nl = lane & 15;
    const int task = blockIdx.x * 8 + wave;      // 528 mtiles * 48 ntiles
    const int mtile = task / 48, n64 = task % 48;

    const bf16* a0 = X  + (size_t)(mtile * 64 + nl) * DIM_;
    const bf16* b0 = WT + (size_t)(n64 * 64 + nl) * DIM_;

    v8f acc[4][4] = {};
    for (int k0 = 0; k0 < DIM_; k0 += 32) {
        v16bf a[4];
#pragma unroll
        for (int i = 0; i < 4; ++i)
            a[i] = load_a_frag(a0 + (size_t)(i * 16) * DIM_, k0, g);
#pragma unroll
        for (int j = 0; j < 4; ++j) {
            v16bf bfr = load_b_frag(b0 + (size_t)(j * 16) * DIM_, k0, g);
#pragma unroll
            for (int i = 0; i < 4; ++i)
                acc[i][j] = wmma_bf16(a[i], bfr, acc[i][j]);
        }
    }
    // Whole 64-col tile lands in one of q/k/v and one head.
    const int n0 = n64 * 64;
    const int which = n0 / 1024;           // 0=q 1=k 2=v
    const int h = (n0 % 1024) / DH_;
    bf16* dst = (which == 0) ? Q : ((which == 1) ? K : V);
    const float scale = (which == 0) ? 0.125f : 1.0f;   // 64^-0.5
    // hoisted batch / seq decomposition (one div per wave, not per element)
    const int m0 = mtile * 64;
    const int bb = m0 / SP_;
    const int s0 = m0 - bb * SP_ + g * 8;
    bf16* drow = dst + ((size_t)(bb * H_ + h) * SP_ + s0) * DH_;
#pragma unroll
    for (int j = 0; j < 4; ++j) {
        int d = j * 16 + nl;
#pragma unroll
        for (int i = 0; i < 4; ++i) {
#pragma unroll
            for (int v = 0; v < 8; ++v)
                drow[(size_t)(i * 16 + v) * DH_ + d] = (bf16)(acc[i][j][v] * scale);
        }
    }
}

// ---------------------------------------------------------------------------
// Stage 3: text attention.  One block per (b,h); 8 waves, wave = 16-row strip.
// S[128,128] = Q_text @ K_text^T (causal), softmax, O = P @ V_text.
// ---------------------------------------------------------------------------
__global__ void __launch_bounds__(256, 1)
attn_text_kernel(const bf16* __restrict__ Q,
                 const bf16* __restrict__ K,
                 const bf16* __restrict__ V,
                 bf16* __restrict__ attn_out) {
    const int bh = blockIdx.x, b = bh / H_, h = bh % H_;
    const size_t base = (size_t)bh * SP_ * DH_;
    const bf16* Qb = Q + base;    // text rows = first 128 positions
    const bf16* Kb = K + base;
    const bf16* Vb = V + base;

    const int wave = threadIdx.x >> 5, lane = threadIdx.x & 31;
    const int g = lane >> 4, nl = lane & 15;

    __shared__ __align__(32) bf16 Pl[TEXT_][TEXT_];   // probs,  32 KB
    __shared__ __align__(32) bf16 Vt[DH_][TEXT_];     // V^T,    16 KB

    for (int idx = threadIdx.x; idx < TEXT_ * DH_; idx += blockDim.x) {
        int s = idx / DH_, d = idx % DH_;
        Vt[d][s] = Vb[(size_t)s * DH_ + d];
    }
    __syncthreads();

    // scores: 16 rows x 128 cols per wave
    v8f acc[8] = {};
    const bf16* arow = Qb + (size_t)(wave * 16 + nl) * DH_;
    for (int k0 = 0; k0 < DH_; k0 += 32) {
        v16bf a = load_a_frag(arow, k0, g);
#pragma unroll
        for (int j = 0; j < 8; ++j) {
            v16bf bfr = load_b_frag(Kb + (size_t)(j * 16 + nl) * DH_, k0, g);
            acc[j] = wmma_bf16(a, bfr, acc[j]);
        }
    }

    // causal mask + softmax (row = acc idx v + 8*g, col = j*16 + nl)
#pragma unroll
    for (int v = 0; v < 8; ++v) {
        const int m = wave * 16 + g * 8 + v;
        float mx = NEGF_;
#pragma unroll
        for (int j = 0; j < 8; ++j) {
            int n = j * 16 + nl;
            float s = acc[j][v];
            if (n > m) s = NEGF_;
            acc[j][v] = s;
            mx = fmaxf(mx, s);
        }
        for (int mk = 1; mk < 16; mk <<= 1) mx = fmaxf(mx, __shfl_xor(mx, mk, 32));
        float sum = 0.f;
#pragma unroll
        for (int j = 0; j < 8; ++j) {
            float e = __expf(acc[j][v] - mx);
            acc[j][v] = e;
            sum += e;
        }
        for (int mk = 1; mk < 16; mk <<= 1) sum += __shfl_xor(sum, mk, 32);
        float inv = 1.f / sum;
#pragma unroll
        for (int j = 0; j < 8; ++j)
            Pl[m][j * 16 + nl] = (bf16)(acc[j][v] * inv);
    }
    __syncthreads();

    // O = P @ V : 16x64 per wave
    v8f oacc[4] = {};
    const bf16* prow = &Pl[wave * 16 + nl][0];
    for (int k0 = 0; k0 < TEXT_; k0 += 32) {
        v16bf a = load_a_frag(prow, k0, g);
#pragma unroll
        for (int j = 0; j < 4; ++j) {
            v16bf bfr = load_b_frag(&Vt[j * 16 + nl][0], k0, g);
            oacc[j] = wmma_bf16(a, bfr, oacc[j]);
        }
    }
    const size_t outb = (size_t)b * SP_ * DIM_;
#pragma unroll
    for (int j = 0; j < 4; ++j) {
        int d = j * 16 + nl;
#pragma unroll
        for (int v = 0; v < 8; ++v) {
            int m = wave * 16 + g * 8 + v;
            attn_out[outb + (size_t)m * DIM_ + h * DH_ + d] = (bf16)oacc[j][v];
        }
    }
}

// ---------------------------------------------------------------------------
// Stage 4: image attention.  One block per (b,h,image-row x): 64 queries
// attend to 128 text keys (mask) + 64 in-row keys (causal). 4 waves/block.
// ---------------------------------------------------------------------------
__global__ void __launch_bounds__(128, 1)
attn_image_kernel(const bf16* __restrict__ Q,
                  const bf16* __restrict__ K,
                  const bf16* __restrict__ V,
                  const unsigned char* __restrict__ mask, // [B][128] bool
                  bf16* __restrict__ attn_out) {
    const int bh = blockIdx.x >> 6, xrow = blockIdx.x & 63;
    const int b = bh / H_, h = bh % H_;
    const size_t base = (size_t)bh * SP_ * DH_;
    const size_t roff = (size_t)(TEXT_ + xrow * IMG_) * DH_;
    const bf16* Qr  = Q + base + roff;     // 64 queries of this image row
    const bf16* Ktx = K + base;            // 128 text keys
    const bf16* Krw = K + base + roff;     // 64 in-row keys
    const bf16* Vtx = V + base;
    const bf16* Vrw = V + base + roff;

    const int wave = threadIdx.x >> 5, lane = threadIdx.x & 31;
    const int g = lane >> 4, nl = lane & 15;

    __shared__ __align__(32) bf16 P[IMG_][192];    // 24 KB
    __shared__ __align__(32) bf16 Vt[DH_][192];    // 24 KB (cols 0-127 text, 128-191 row)

    for (int idx = threadIdx.x; idx < DH_ * 192; idx += blockDim.x) {
        int d = idx / 192, c = idx % 192;
        Vt[d][c] = (c < TEXT_) ? Vtx[(size_t)c * DH_ + d]
                               : Vrw[(size_t)(c - TEXT_) * DH_ + d];
    }
    __syncthreads();

    // scores: 16 rows x 192 cols per wave (12 f32 accumulators)
    v8f acc[12] = {};
    const bf16* arow = Qr + (size_t)(wave * 16 + nl) * DH_;
    for (int k0 = 0; k0 < DH_; k0 += 32) {
        v16bf a = load_a_frag(arow, k0, g);
#pragma unroll
        for (int j = 0; j < 8; ++j) {
            v16bf bfr = load_b_frag(Ktx + (size_t)(j * 16 + nl) * DH_, k0, g);
            acc[j] = wmma_bf16(a, bfr, acc[j]);
        }
#pragma unroll
        for (int j = 0; j < 4; ++j) {
            v16bf bfr = load_b_frag(Krw + (size_t)(j * 16 + nl) * DH_, k0, g);
            acc[8 + j] = wmma_bf16(a, bfr, acc[8 + j]);
        }
    }

    // mask + softmax over 192 keys
#pragma unroll
    for (int v = 0; v < 8; ++v) {
        const int i = wave * 16 + g * 8 + v;     // query index within row (0..63)
        float mx = NEGF_;
#pragma unroll
        for (int j = 0; j < 12; ++j) {
            float s = acc[j][v];
            if (j < 8) {
                int n = j * 16 + nl;
                if (!mask[b * TEXT_ + n]) s = NEGF_;
            } else {
                int n = (j - 8) * 16 + nl;
                if (n > i) s = NEGF_;            // causal within image row
            }
            acc[j][v] = s;
            mx = fmaxf(mx, s);
        }
        for (int mk = 1; mk < 16; mk <<= 1) mx = fmaxf(mx, __shfl_xor(mx, mk, 32));
        float sum = 0.f;
#pragma unroll
        for (int j = 0; j < 12; ++j) {
            float e = __expf(acc[j][v] - mx);
            acc[j][v] = e;
            sum += e;
        }
        for (int mk = 1; mk < 16; mk <<= 1) sum += __shfl_xor(sum, mk, 32);
        float inv = 1.f / sum;
#pragma unroll
        for (int j = 0; j < 12; ++j)
            P[i][j * 16 + nl] = (bf16)(acc[j][v] * inv);
    }
    __syncthreads();

    // O = P[64,192] @ V[192,64]
    v8f oacc[4] = {};
    const bf16* prow = &P[wave * 16 + nl][0];
    for (int k0 = 0; k0 < 192; k0 += 32) {
        v16bf a = load_a_frag(prow, k0, g);
#pragma unroll
        for (int j = 0; j < 4; ++j) {
            v16bf bfr = load_b_frag(&Vt[j * 16 + nl][0], k0, g);
            oacc[j] = wmma_bf16(a, bfr, oacc[j]);
        }
    }
    const size_t srow0 = (size_t)b * SP_ + TEXT_ + xrow * IMG_;
#pragma unroll
    for (int j = 0; j < 4; ++j) {
        int d = j * 16 + nl;
#pragma unroll
        for (int v = 0; v < 8; ++v) {
            int i = wave * 16 + g * 8 + v;
            attn_out[(srow0 + i) * DIM_ + h * DH_ + d] = (bf16)oacc[j][v];
        }
    }
}

// ---------------------------------------------------------------------------
// Stage 5: output projection.  out[b,0..4222,1024] = AttnOut @ w_out (f32 out).
// 64x64 wave tile, same structure as the QKV GEMM; hoisted b/s epilogue.
// ---------------------------------------------------------------------------
__global__ void __launch_bounds__(256, 1)
out_gemm_kernel(const bf16* __restrict__ A,    // [33792][1024]
                const bf16* __restrict__ WT,   // [1024][1024] = w_out^T
                float* __restrict__ out) {
    const int wave = threadIdx.x >> 5, lane = threadIdx.x & 31;
    const int g = lane >> 4, nl = lane & 15;
    const int task = blockIdx.x * 8 + wave;      // 528 mtiles * 16 ntiles
    const int mtile = task / 16, n64 = task % 16;

    const bf16* a0 = A  + (size_t)(mtile * 64 + nl) * DIM_;
    const bf16* b0 = WT + (size_t)(n64 * 64 + nl) * DIM_;

    v8f acc[4][4] = {};
    for (int k0 = 0; k0 < DIM_; k0 += 32) {
        v16bf a[4];
#pragma unroll
        for (int i = 0; i < 4; ++i)
            a[i] = load_a_frag(a0 + (size_t)(i * 16) * DIM_, k0, g);
#pragma unroll
        for (int j = 0; j < 4; ++j) {
            v16bf bfr = load_b_frag(b0 + (size_t)(j * 16) * DIM_, k0, g);
#pragma unroll
            for (int i = 0; i < 4; ++i)
                acc[i][j] = wmma_bf16(a[i], bfr, acc[i][j]);
        }
    }
    // hoisted batch / seq decomposition (tile never crosses batch boundary)
    const int m0 = mtile * 64;
    const int bb = m0 / SP_;
    const int s0 = m0 - bb * SP_ + g * 8;
    float* orow = out + ((size_t)bb * SEQ_ + s0) * DIM_;
#pragma unroll
    for (int j = 0; j < 4; ++j) {
        int n = n64 * 64 + j * 16 + nl;
#pragma unroll
        for (int i = 0; i < 4; ++i) {
#pragma unroll
            for (int v = 0; v < 8; ++v) {
                if (s0 + i * 16 + v < SEQ_)     // drop pad row
                    orow[(size_t)(i * 16 + v) * DIM_ + n] = acc[i][j][v];
            }
        }
    }
}

// ---------------------------------------------------------------------------
// Launch.  Workspace layout (bytes, all 256B-multiples), ~285 MB total:
//   Xbf   : ROWS_*DIM_*2            = 69,206,016   (reused as attn_out later)
//   WqkvT : 3072*1024*2             =  6,291,456
//   WoutT : 1024*1024*2             =  2,097,152
//   Q,K,V : 3 * 69,206,016
// ---------------------------------------------------------------------------
extern "C" void kernel_launch(void* const* d_in, const int* in_sizes, int n_in,
                              void* d_out, int out_size, void* d_ws, size_t ws_size,
                              hipStream_t stream) {
    const float*         x     = (const float*)d_in[0];
    const unsigned char* mask  = (const unsigned char*)d_in[1];  // jnp bool -> bytes
    const float*         wqkv  = (const float*)d_in[2];
    const float*         wout  = (const float*)d_in[3];
    float*               out   = (float*)d_out;

    const size_t SZ_X = (size_t)ROWS_ * DIM_ * sizeof(bf16);     // 69,206,016
    char* ws = (char*)d_ws;
    bf16* Xbf   = (bf16*)ws;                 ws += SZ_X;
    bf16* WqkvT = (bf16*)ws;                 ws += (size_t)3072 * 1024 * sizeof(bf16);
    bf16* WoutT = (bf16*)ws;                 ws += (size_t)1024 * 1024 * sizeof(bf16);
    bf16* Qh    = (bf16*)ws;                 ws += SZ_X;
    bf16* Kh    = (bf16*)ws;                 ws += SZ_X;
    bf16* Vh    = (bf16*)ws;                 ws += SZ_X;
    bf16* AttnO = Xbf;   // alias: X is dead after the QKV GEMM (stream-ordered)

    // Stage 1: convert / transpose
    {
        size_t nx = (size_t)ROWS_ * DIM_;
        cvt_x_kernel<<<(unsigned)((nx + 255) / 256), 256, 0, stream>>>(x, Xbf);
        size_t nq = (size_t)DIM_ * 3072;
        cvt_wT_kernel<<<(unsigned)((nq + 255) / 256), 256, 0, stream>>>(wqkv, WqkvT, DIM_, 3072);
        size_t no = (size_t)DIM_ * DIM_;
        cvt_wT_kernel<<<(unsigned)((no + 255) / 256), 256, 0, stream>>>(wout, WoutT, DIM_, DIM_);
    }
    // Stage 2: QKV projection — 528 mtiles * 48 ntiles / 8 waves = 3168 blocks
    qkv_gemm_kernel<<<3168, 256, 0, stream>>>(Xbf, WqkvT, Qh, Kh, Vh);
    // Stage 3: text attention — one block per (b,h)
    attn_text_kernel<<<B_ * H_, 256, 0, stream>>>(Qh, Kh, Vh, AttnO);
    // Stage 4: image attention — one block per (b,h,row)
    attn_image_kernel<<<B_ * H_ * IMG_, 128, 0, stream>>>(Qh, Kh, Vh, mask, AttnO);
    // Stage 5: output projection — 528 mtiles * 16 ntiles / 8 waves = 1056 blocks
    out_gemm_kernel<<<1056, 256, 0, stream>>>(AttnO, WoutT, out);
}
